// VAERouter_2190433321663
// MI455X (gfx1250) — matrix operand order repaired and checked by reading, not compiled, not run on hardware
//
#include <hip/hip_runtime.h>
#include <hip/hip_bf16.h>

typedef __attribute__((ext_vector_type(16))) _Float16 v16h;
typedef __attribute__((ext_vector_type(8)))  float    v8f;

#define B_TOT 16384
#define A_N   32
#define SLOT  64
#define H1D   256
#define H2D   128
#define L_DIM 32

union FragU { v16h v; uint4 q[2]; };

// A-fragment (16x32 f16, M x K) from a row-major [16 x ld] block.
// ISA layout: lanes 0-15: M=lane, halves 0-7 = K 0..7, halves 8-15 = K 16..23;
//             lanes 16-31: K 8..15 and K 24..31.  -> two 16B chunks per lane.
__device__ __forceinline__ v16h load_a_frag(const _Float16* blk, int ld, int kt) {
  const int lane = threadIdx.x & 31;
  const int m    = lane & 15;
  const int hi   = lane >> 4;
  const _Float16* p = blk + m * ld + kt * 32 + hi * 8;
  FragU u;
  u.q[0] = *(const uint4*)(p);
  u.q[1] = *(const uint4*)(p + 16);
  return u.v;
}

// B-fragment (32x16 f16, K x N) from weight matrix stored as rows of N,
// contiguous in K (i.e. W[n][k], ld = K-stride of a row). ISA layout:
// lane = n + 16*hi, halves 0-7 = K 8hi..8hi+7, halves 8-15 = K 16+8hi..16+8hi+7.
__device__ __forceinline__ v16h load_b_frag(const _Float16* w, int ld, int nt, int kt) {
  const int lane = threadIdx.x & 31;
  const int n    = lane & 15;
  const int hi   = lane >> 4;
  const _Float16* p = w + (nt * 16 + n) * ld + kt * 32 + hi * 8;
  FragU u;
  u.q[0] = *(const uint4*)(p);
  u.q[1] = *(const uint4*)(p + 16);
  return u.v;
}

__global__ void cvt_f32_to_f16(const float* __restrict__ in,
                               _Float16* __restrict__ out, int n) {
  int i = blockIdx.x * blockDim.x + threadIdx.x;
  int stride = gridDim.x * blockDim.x;
  for (; i < n; i += stride) out[i] = (_Float16)in[i];
}

// One wave handles a 32-row batch tile (two 16x16 M-tiles sharing each
// B-fragment -> 2x weight reuse) for one agent. A-fragments are hoisted into
// registers across the N-loops (LDS-bound occupancy leaves VGPRs free).
// 2 waves / block, 60KB static LDS.
__global__ __launch_bounds__(64) void vae_wmma_kernel(
    const _Float16* __restrict__ sF,  const _Float16* __restrict__ W1f,
    const _Float16* __restrict__ W2f, const _Float16* __restrict__ Wmf,
    const _Float16* __restrict__ Sf,
    const float* __restrict__ b1, const float* __restrict__ b2,
    const float* __restrict__ bmu, const float* __restrict__ mus,
    float* __restrict__ scores) {
  __shared__ __align__(16) _Float16 h1s[2][32 * H1D];    // 2 x 16KB
  __shared__ __align__(16) _Float16 h2s[2][32 * H2D];    // 2 x  8KB
  __shared__ __align__(16) _Float16 d16s[2][32 * L_DIM]; // 2 x  2KB
  __shared__ __align__(16) float    d32s[2][32 * L_DIM]; // 2 x  4KB

  const int wave = threadIdx.x >> 5;
  const int lane = threadIdx.x & 31;
  const int n    = lane & 15;
  const int hi   = lane >> 4;

  const int wid   = blockIdx.x * 2 + wave;   // 16384 waves total
  const int agent = wid >> 9;                // 512 tiles per agent
  const int tile  = wid & 511;
  const int b0    = tile * 32;

  _Float16* h1w  = h1s[wave];
  _Float16* h2w  = h2s[wave];
  _Float16* d16w = d16s[wave];
  float*    d32w = d32s[wave];

  // ---------- Layer 1: h1 = relu(s @ W1^T + b1)  (32x64)x(64x256) ----------
  const _Float16* sblk = sF + (size_t)b0 * SLOT;
  const _Float16* W1a  = W1f + agent * (H1D * SLOT);
  {
    v16h aS[2][2];                       // whole 32x64 s-tile: 32 VGPRs
#pragma unroll
    for (int mt = 0; mt < 2; ++mt)
#pragma unroll
      for (int kt = 0; kt < 2; ++kt)
        aS[mt][kt] = load_a_frag(sblk + mt * 16 * SLOT, SLOT, kt);

    for (int nt = 0; nt < 16; ++nt) {
      v8f acc0 = {}, acc1 = {};
#pragma unroll
      for (int kt = 0; kt < 2; ++kt) {
        v16h bv = load_b_frag(W1a, SLOT, nt, kt);
        acc0 = __builtin_amdgcn_wmma_f32_16x16x32_f16(false, aS[0][kt], false, bv,
                                                      (short)0, acc0, false, false);
        acc1 = __builtin_amdgcn_wmma_f32_16x16x32_f16(false, aS[1][kt], false, bv,
                                                      (short)0, acc1, false, false);
      }
      float bias = b1[agent * H1D + nt * 16 + n];
#pragma unroll
      for (int r = 0; r < 8; ++r) {
        float v0 = acc0[r] + bias;
        float v1 = acc1[r] + bias;
        v0 = v0 > 0.f ? v0 : 0.f;
        v1 = v1 > 0.f ? v1 : 0.f;
        h1w[(r + hi * 8) * H1D + nt * 16 + n]      = (_Float16)v0;
        h1w[(16 + r + hi * 8) * H1D + nt * 16 + n] = (_Float16)v1;
      }
    }
  }
  asm volatile("s_wait_dscnt 0" ::: "memory");

  // ---------- Layer 2: h2 = relu(h1 @ W2^T + b2)  (32x256)x(256x128) ----------
  const _Float16* W2a = W2f + agent * (H2D * H1D);
  {
    v16h a1[2][8];                       // whole 32x256 h1 tile: 128 VGPRs
#pragma unroll
    for (int mt = 0; mt < 2; ++mt)
#pragma unroll
      for (int kt = 0; kt < 8; ++kt)
        a1[mt][kt] = load_a_frag(h1w + mt * 16 * H1D, H1D, kt);

    for (int nt = 0; nt < 8; ++nt) {
      v8f acc0 = {}, acc1 = {};
#pragma unroll
      for (int kt = 0; kt < 8; ++kt) {
        v16h bv = load_b_frag(W2a, H1D, nt, kt);
        acc0 = __builtin_amdgcn_wmma_f32_16x16x32_f16(false, a1[0][kt], false, bv,
                                                      (short)0, acc0, false, false);
        acc1 = __builtin_amdgcn_wmma_f32_16x16x32_f16(false, a1[1][kt], false, bv,
                                                      (short)0, acc1, false, false);
      }
      float bias = b2[agent * H2D + nt * 16 + n];
#pragma unroll
      for (int r = 0; r < 8; ++r) {
        float v0 = acc0[r] + bias;
        float v1 = acc1[r] + bias;
        v0 = v0 > 0.f ? v0 : 0.f;
        v1 = v1 > 0.f ? v1 : 0.f;
        h2w[(r + hi * 8) * H2D + nt * 16 + n]      = (_Float16)v0;
        h2w[(16 + r + hi * 8) * H2D + nt * 16 + n] = (_Float16)v1;
      }
    }
  }
  asm volatile("s_wait_dscnt 0" ::: "memory");

  // ---------- Layer 3: diff = h2 @ Wmu^T + bmu - mu_stats  (32x128)x(128x32) ----------
  const _Float16* Wma = Wmf + agent * (L_DIM * H2D);
  {
    v16h a2[2][4];                       // whole 32x128 h2 tile: 64 VGPRs
#pragma unroll
    for (int mt = 0; mt < 2; ++mt)
#pragma unroll
      for (int kt = 0; kt < 4; ++kt)
        a2[mt][kt] = load_a_frag(h2w + mt * 16 * H2D, H2D, kt);

#pragma unroll
    for (int nt = 0; nt < 2; ++nt) {
      v8f acc0 = {}, acc1 = {};
#pragma unroll
      for (int kt = 0; kt < 4; ++kt) {
        v16h bv = load_b_frag(Wma, H2D, nt, kt);
        acc0 = __builtin_amdgcn_wmma_f32_16x16x32_f16(false, a2[0][kt], false, bv,
                                                      (short)0, acc0, false, false);
        acc1 = __builtin_amdgcn_wmma_f32_16x16x32_f16(false, a2[1][kt], false, bv,
                                                      (short)0, acc1, false, false);
      }
      int col = nt * 16 + n;
      float add = bmu[agent * L_DIM + col] - mus[agent * L_DIM + col];
#pragma unroll
      for (int r = 0; r < 8; ++r) {
        float v0 = acc0[r] + add;
        float v1 = acc1[r] + add;
        int m = r + hi * 8;
        d16w[m * L_DIM + col]        = (_Float16)v0;
        d32w[m * L_DIM + col]        = v0;
        d16w[(16 + m) * L_DIM + col] = (_Float16)v1;
        d32w[(16 + m) * L_DIM + col] = v1;
      }
    }
  }
  asm volatile("s_wait_dscnt 0" ::: "memory");

  // ---------- Layer 4: left = diff @ Sigma_inv (symmetric -> row-major B),
  //            score = -sum(left * diff) ----------
  const _Float16* Sa = Sf + agent * (L_DIM * L_DIM);
  float ps[2][8];
#pragma unroll
  for (int mt = 0; mt < 2; ++mt)
#pragma unroll
    for (int r = 0; r < 8; ++r) ps[mt][r] = 0.f;

  {
    v16h aD[2];
#pragma unroll
    for (int mt = 0; mt < 2; ++mt)
      aD[mt] = load_a_frag(d16w + mt * 16 * L_DIM, L_DIM, 0);

#pragma unroll
    for (int nt = 0; nt < 2; ++nt) {
      v16h bv = load_b_frag(Sa, L_DIM, nt, 0);
#pragma unroll
      for (int mt = 0; mt < 2; ++mt) {
        v8f acc = {};
        acc = __builtin_amdgcn_wmma_f32_16x16x32_f16(false, aD[mt], false, bv,
                                                     (short)0, acc, false, false);
#pragma unroll
        for (int r = 0; r < 8; ++r) {
          int m = mt * 16 + r + hi * 8;
          ps[mt][r] += acc[r] * d32w[m * L_DIM + nt * 16 + n];
        }
      }
    }
  }
  // Reduce over N (lanes 0-15 carry M=0..7, lanes 16-31 carry M=8..15);
  // xor masks < 16 keep the two halves separate.
#pragma unroll
  for (int mt = 0; mt < 2; ++mt) {
#pragma unroll
    for (int r = 0; r < 8; ++r) {
      float v = ps[mt][r];
      v += __shfl_xor(v, 1, 32);
      v += __shfl_xor(v, 2, 32);
      v += __shfl_xor(v, 4, 32);
      v += __shfl_xor(v, 8, 32);
      ps[mt][r] = v;
    }
  }
  if (n == 0) {  // lane 0 (M=0..7) and lane 16 (M=8..15) of each M-tile
#pragma unroll
    for (int mt = 0; mt < 2; ++mt)
#pragma unroll
      for (int r = 0; r < 8; ++r)
        scores[(size_t)(b0 + mt * 16 + hi * 8 + r) * A_N + agent] = -ps[mt][r];
  }
}

__global__ void route_kernel(const float* __restrict__ scores,
                             int* __restrict__ assigned,
                             const float* __restrict__ theta) {
  int b = blockIdx.x * blockDim.x + threadIdx.x;
  if (b >= B_TOT) return;
  const float* row = scores + (size_t)b * A_N;
  float best = row[0];
  int bi = 0;
#pragma unroll
  for (int a = 1; a < A_N; ++a) {
    float v = row[a];
    if (v > best) { best = v; bi = a; }
  }
  assigned[b] = (best >= theta[0]) ? bi : -1;
}

extern "C" void kernel_launch(void* const* d_in, const int* in_sizes, int n_in,
                              void* d_out, int out_size, void* d_ws, size_t ws_size,
                              hipStream_t stream) {
  const float* s     = (const float*)d_in[0];
  const float* W1    = (const float*)d_in[1];
  const float* b1    = (const float*)d_in[2];
  const float* W2    = (const float*)d_in[3];
  const float* b2    = (const float*)d_in[4];
  const float* Wmu   = (const float*)d_in[5];
  const float* bmu   = (const float*)d_in[6];
  const float* mus   = (const float*)d_in[7];
  const float* Sigma = (const float*)d_in[8];
  const float* theta = (const float*)d_in[9];

  // f16 staging in workspace (~5.4 MB total)
  _Float16* sF  = (_Float16*)d_ws;
  _Float16* W1f = sF  + (size_t)B_TOT * SLOT;        // 1,048,576 halfs
  _Float16* W2f = W1f + (size_t)A_N * H1D * SLOT;    //   524,288
  _Float16* Wmf = W2f + (size_t)A_N * H2D * H1D;     // 1,048,576
  _Float16* Sf  = Wmf + (size_t)A_N * L_DIM * H2D;   //   131,072
                                                     // Sf: 32,768

  cvt_f32_to_f16<<<2048, 256, 0, stream>>>(s,     sF,  B_TOT * SLOT);
  cvt_f32_to_f16<<<1024, 256, 0, stream>>>(W1,    W1f, A_N * H1D * SLOT);
  cvt_f32_to_f16<<<2048, 256, 0, stream>>>(W2,    W2f, A_N * H2D * H1D);
  cvt_f32_to_f16<<<256,  256, 0, stream>>>(Wmu,   Wmf, A_N * L_DIM * H2D);
  cvt_f32_to_f16<<<64,   256, 0, stream>>>(Sigma, Sf,  A_N * L_DIM * L_DIM);

  float* scores  = (float*)d_out;
  int* assigned  = (int*)(scores + (size_t)B_TOT * A_N);

  // 16384 waves = (B/32 tiles) x (A agents); 2 waves (one agent) per block
  vae_wmma_kernel<<<8192, 64, 0, stream>>>(sF, W1f, W2f, Wmf, Sf,
                                           b1, b2, bmu, mus, scores);
  route_kernel<<<(B_TOT + 255) / 256, 256, 0, stream>>>(scores, assigned, theta);
}
